// GATModel_16776142258484
// MI455X (gfx1250) — matrix operand order repaired and checked by reading, not compiled, not run on hardware
//
#include <hip/hip_runtime.h>
#include <hip/hip_bf16.h>

#define N_NODES 50000
#define N_EDGES 800000
#define NTOT    (N_EDGES + N_NODES)   // edges + self loops
#define DIM     128                   // D_IN == H*C
#define NHEAD   4
#define CHAN    32
#define NQ      10000

typedef __attribute__((ext_vector_type(16))) _Float16 v16h;
typedef __attribute__((ext_vector_type(8)))  _Float16 v8h;
typedef __attribute__((ext_vector_type(8)))  float    v8f;

// ---------------- helpers ----------------
__device__ __forceinline__ unsigned f32_enc(float f) {
  unsigned b = __float_as_uint(f);
  return (b & 0x80000000u) ? ~b : (b | 0x80000000u);   // monotone map f32 -> u32
}
__device__ __forceinline__ float f32_dec(unsigned u) {
  unsigned b = (u & 0x80000000u) ? (u & 0x7FFFFFFFu) : ~u;
  return __uint_as_float(b);
}
__device__ __forceinline__ void edge_sd(const int* __restrict__ ei, int i, int& s, int& d) {
  if (i < N_EDGES) { s = ei[i]; d = ei[N_EDGES + i]; }
  else             { s = i - N_EDGES; d = s; }          // self loops
}

// ---------------- fp32 -> fp16 convert (optional bias+relu fold) ----------------
__global__ void k_cvt_f16(const float* __restrict__ src, _Float16* __restrict__ dst,
                          const float* __restrict__ bias, int relu, int count) {
  int i = blockIdx.x * blockDim.x + threadIdx.x;
  if (i >= count) return;
  float v = src[i];
  if (bias) v += bias[i & (DIM - 1)];
  if (relu) v = v > 0.f ? v : 0.f;
  dst[i] = (_Float16)v;
}

// ---------------- W[k][n] f32 -> Wt[n][k] f16 (column-major B for WMMA) ----------------
__global__ void k_wt_f16(const float* __restrict__ W, _Float16* __restrict__ Wt) {
  int t = blockIdx.x * blockDim.x + threadIdx.x;   // 0..16383
  int n = t >> 7, k = t & 127;
  Wt[t] = (_Float16)W[k * DIM + n];
}

// ---------------- WMMA GEMM: D[M,128] = A[M,128](f16) @ W(Bt is [n][k] f16) (+bias) ----------------
// one wave = one 16x16 tile; 8 waves/block cover the 128 output columns.
__global__ void k_wmma_gemm(const _Float16* __restrict__ A, const _Float16* __restrict__ Bt,
                            const float* __restrict__ bias, float* __restrict__ D) {
  int lane = threadIdx.x & 31;
  int wave = threadIdx.x >> 5;         // N-tile index 0..7
  int row0 = blockIdx.x << 4;
  int col0 = wave << 4;
  int l15  = lane & 15;
  int ksel = lane >> 4;                // half-wave select

  // A 16x32 f16, contiguous k order chosen here (same as B): lanes 0-15 M=0..15 hold K 0..15; lanes 16-31 K 16..31
  const _Float16* aptr = A  + (size_t)(row0 + l15) * DIM + ksel * 16;
  // B 32x16 f16, same k order as A: lane holds one column, contiguous 16 K values per K-step
  const _Float16* bptr = Bt + (size_t)(col0 + l15) * DIM + ksel * 16;

  v8f c = {};
#pragma unroll
  for (int k0 = 0; k0 < DIM; k0 += 32) {
    v8h alo = *(const v8h*)(aptr + k0);
    v8h ahi = *(const v8h*)(aptr + k0 + 8);
    v16h a = __builtin_shufflevector(alo, ahi, 0,1,2,3,4,5,6,7,8,9,10,11,12,13,14,15);
    v16h b = *(const v16h*)(bptr + k0);
    c = __builtin_amdgcn_wmma_f32_16x16x32_f16(false, a, false, b, (short)0, c, false, false);
  }
  // C/D layout: lane -> column col0+l15; VGPR r -> row row0 + r + 8*ksel
  int ccol  = col0 + l15;
  int rbase = row0 + (ksel << 3);
  float badd = bias ? bias[ccol] : 0.f;
#pragma unroll
  for (int r = 0; r < 8; ++r)
    D[(size_t)(rbase + r) * DIM + ccol] = c[r] + badd;
}

// ---------------- attention coefficients: alpha_s/d[n,h] = sum_c h[n,h,c]*a[h,c] ----------------
__global__ void k_attn(const float* __restrict__ h, const float* __restrict__ asrc,
                       const float* __restrict__ adst, float* __restrict__ als,
                       float* __restrict__ ald) {
  int n    = blockIdx.x;
  int head = threadIdx.x >> 5;     // one wave32 per head
  int lane = threadIdx.x & 31;
  float v = h[(size_t)n * DIM + head * CHAN + lane];
  float s = v * asrc[head * CHAN + lane];
  float d = v * adst[head * CHAN + lane];
#pragma unroll
  for (int off = 16; off >= 1; off >>= 1) {
    s += __shfl_xor(s, off, 32);
    d += __shfl_xor(d, off, 32);
  }
  if (lane == 0) { als[n * NHEAD + head] = s; ald[n * NHEAD + head] = d; }
}

// ---------------- per-layer scratch init ----------------
__global__ void k_init(unsigned* __restrict__ menc, float* __restrict__ z,
                       float* __restrict__ out) {
  int i = blockIdx.x * blockDim.x + threadIdx.x;
  if (i < N_NODES * NHEAD) { menc[i] = 0u; z[i] = 0.f; }   // enc(-inf) < any enc(finite)
  if (i < N_NODES * DIM)   out[i] = 0.f;
}

// ---------------- edge pass 1: leaky-relu logits + segment max ----------------
__global__ void k_edge_max(const int* __restrict__ ei, const float* __restrict__ als,
                           const float* __restrict__ ald, float* __restrict__ ework,
                           unsigned* __restrict__ menc) {
  int t = blockIdx.x * blockDim.x + threadIdx.x;
  if (t >= NTOT * NHEAD) return;
  int i = t >> 2, hd = t & 3;
  int s, d; edge_sd(ei, i, s, d);
  float e = als[s * NHEAD + hd] + ald[d * NHEAD + hd];
  e = e > 0.f ? e : 0.2f * e;            // leaky relu, slope 0.2
  ework[t] = e;
  atomicMax(&menc[d * NHEAD + hd], f32_enc(e));
}

// ---------------- edge pass 2: exp + segment sum ----------------
__global__ void k_edge_exp(const int* __restrict__ ei, const unsigned* __restrict__ menc,
                           float* __restrict__ ework, float* __restrict__ z) {
  int t = blockIdx.x * blockDim.x + threadIdx.x;
  if (t >= NTOT * NHEAD) return;
  int i = t >> 2, hd = t & 3;
  int s, d; edge_sd(ei, i, s, d);
  float m  = f32_dec(menc[d * NHEAD + hd]);
  float ex = __expf(ework[t] - m);
  ework[t] = ex;
  atomicAdd(&z[d * NHEAD + hd], ex);
}

// ---------------- edge pass 3: weighted aggregate (one 128-thread block per edge) ----------------
__global__ void k_aggregate(const int* __restrict__ ei, const float* __restrict__ hbuf,
                            const float* __restrict__ ework, const float* __restrict__ z,
                            float* __restrict__ out) {
  int i = blockIdx.x;          // edge
  int c = threadIdx.x;         // feature 0..127
  int hd = c >> 5;
  int s, d; edge_sd(ei, i, s, d);
  float alpha = ework[i * NHEAD + hd] / z[d * NHEAD + hd];
  atomicAdd(&out[(size_t)d * DIM + c], hbuf[(size_t)s * DIM + c] * alpha);
}

// ---------------- final h output: out + bias -> d_out ----------------
__global__ void k_bias_store(const float* __restrict__ src, const float* __restrict__ bias,
                             float* __restrict__ dst, int count) {
  int i = blockIdx.x * blockDim.x + threadIdx.x;
  if (i >= count) return;
  dst[i] = src[i] + bias[i & (DIM - 1)];
}

extern "C" void kernel_launch(void* const* d_in, const int* in_sizes, int n_in,
                              void* d_out, int out_size, void* d_ws, size_t ws_size,
                              hipStream_t stream) {
  const float* x   = (const float*)d_in[0];
  const int*   ei  = (const int*)  d_in[1];
  const float* W1  = (const float*)d_in[2];
  const float* as1 = (const float*)d_in[3];
  const float* ad1 = (const float*)d_in[4];
  const float* b1  = (const float*)d_in[5];
  const float* W2  = (const float*)d_in[6];
  const float* as2 = (const float*)d_in[7];
  const float* ad2 = (const float*)d_in[8];
  const float* b2  = (const float*)d_in[9];
  const float* qe  = (const float*)d_in[10];
  const float* qW  = (const float*)d_in[11];
  const float* qb  = (const float*)d_in[12];

  float* out_ques = (float*)d_out;                          // [NQ,128]
  float* out_h    = (float*)d_out + (size_t)NQ * DIM;       // [N,128]

  // ---- workspace carve (all 256B aligned) ----
  char* p = (char*)d_ws;
  auto take = [&](size_t bytes) { char* r = p; p += (bytes + 255) & ~(size_t)255; return r; };
  _Float16* xh16 = (_Float16*)take((size_t)N_NODES * DIM * sizeof(_Float16));
  _Float16* Wt16 = (_Float16*)take((size_t)DIM * DIM * sizeof(_Float16));
  float*    hbuf = (float*)   take((size_t)N_NODES * DIM * sizeof(float));
  float*    outb = (float*)   take((size_t)N_NODES * DIM * sizeof(float));
  float*    als  = (float*)   take((size_t)N_NODES * NHEAD * sizeof(float));
  float*    ald  = (float*)   take((size_t)N_NODES * NHEAD * sizeof(float));
  unsigned* menc = (unsigned*)take((size_t)N_NODES * NHEAD * sizeof(unsigned));
  float*    z    = (float*)   take((size_t)N_NODES * NHEAD * sizeof(float));
  float*    ewk  = (float*)   take((size_t)NTOT * NHEAD * sizeof(float));

  const int T = 256;
  auto cdiv = [](long a, long b) { return (int)((a + b - 1) / b); };
  const int gND  = cdiv((long)N_NODES * DIM, T);
  const int gEH  = cdiv((long)NTOT * NHEAD, T);

  // ================= layer 1 =================
  k_cvt_f16   <<<gND, T, 0, stream>>>(x, xh16, nullptr, 0, N_NODES * DIM);
  k_wt_f16    <<<cdiv(DIM * DIM, T), T, 0, stream>>>(W1, Wt16);
  k_wmma_gemm <<<N_NODES / 16, 256, 0, stream>>>(xh16, Wt16, nullptr, hbuf);
  k_attn      <<<N_NODES, 128, 0, stream>>>(hbuf, as1, ad1, als, ald);
  k_init      <<<gND, T, 0, stream>>>(menc, z, outb);
  k_edge_max  <<<gEH, T, 0, stream>>>(ei, als, ald, ewk, menc);
  k_edge_exp  <<<gEH, T, 0, stream>>>(ei, menc, ewk, z);
  k_aggregate <<<NTOT, DIM, 0, stream>>>(ei, hbuf, ewk, z, outb);
  // layer-2 input = relu(out + bias1), cast to f16
  k_cvt_f16   <<<gND, T, 0, stream>>>(outb, xh16, b1, 1, N_NODES * DIM);

  // ================= layer 2 =================
  k_wt_f16    <<<cdiv(DIM * DIM, T), T, 0, stream>>>(W2, Wt16);
  k_wmma_gemm <<<N_NODES / 16, 256, 0, stream>>>(xh16, Wt16, nullptr, hbuf);
  k_attn      <<<N_NODES, 128, 0, stream>>>(hbuf, as2, ad2, als, ald);
  k_init      <<<gND, T, 0, stream>>>(menc, z, outb);
  k_edge_max  <<<gEH, T, 0, stream>>>(ei, als, ald, ewk, menc);
  k_edge_exp  <<<gEH, T, 0, stream>>>(ei, menc, ewk, z);
  k_aggregate <<<NTOT, DIM, 0, stream>>>(ei, hbuf, ewk, z, outb);
  k_bias_store<<<gND, T, 0, stream>>>(outb, b2, out_h, N_NODES * DIM);

  // ================= query head =================
  k_cvt_f16   <<<cdiv((long)NQ * DIM, T), T, 0, stream>>>(qe, xh16, nullptr, 0, NQ * DIM);
  k_wt_f16    <<<cdiv(DIM * DIM, T), T, 0, stream>>>(qW, Wt16);
  k_wmma_gemm <<<NQ / 16, 256, 0, stream>>>(xh16, Wt16, qb, out_ques);
}
